// CausalSelfAttention_33200097198695
// MI455X (gfx1250) — matrix operand order repaired and checked by reading, not compiled, not run on hardware
//
#include <hip/hip_runtime.h>
#include <hip/hip_bf16.h>

// ---------------------------------------------------------------------------
// Causal self-attention forward for MI455X (gfx1250, wave32, WMMA).
//   B=4, T=2048, D=1024, H=16, HD=64
// Pipeline: qkv = x@Wqkv -> bf16 Q/K/V per head  (WMMA bf16 GEMM, dbl-buffered)
//           flash-attention per (b,h,128-row q tile)  (WMMA bf16, async K tiles)
//           out = yatt@Wout + bout                 (WMMA bf16 GEMM, async A tiles)
// ---------------------------------------------------------------------------

#define B_  4
#define T_  2048
#define D_  1024
#define H_  16
#define HD_ 64
#define NEG_ (-1.0e9f)

typedef __attribute__((ext_vector_type(16))) __bf16 bf16x16;
typedef __attribute__((ext_vector_type(8)))  float  f32x8;
typedef __attribute__((ext_vector_type(4)))  int    i32x4;

union Frag16 {                 // one WMMA 16-bit A/B fragment (8 VGPRs)
  bf16x16        v;
  uint4          q[2];
  unsigned short u[16];
};

union U8x16 {                  // helper for unpacking a 16B vector load
  uint4          q;
  unsigned short u[8];
};

__device__ __forceinline__ unsigned short f2bf(float f) {
  union { float f; unsigned int u; } cv; cv.f = f;
  unsigned int u = cv.u;
  return (unsigned short)((u + 0x7FFFu + ((u >> 16) & 1u)) >> 16);  // RNE
}

__device__ __forceinline__ f32x8 zero8() {
  f32x8 z;
#pragma unroll
  for (int i = 0; i < 8; ++i) z[i] = 0.0f;
  return z;
}

// ---- CDNA5 async global->LDS copy (guarded: falls back to sync copy) ------
#if defined(__gfx1250__) && \
    __has_builtin(__builtin_amdgcn_global_load_async_to_lds_b128) && \
    __has_builtin(__builtin_amdgcn_s_wait_asynccnt)
#define USE_ASYNC_LDS 1
#else
#define USE_ASYNC_LDS 0
#endif

__device__ __forceinline__ void copy_b128_to_lds(const unsigned short* gsrc,
                                                 unsigned short* ldst) {
#if USE_ASYNC_LDS
  __builtin_amdgcn_global_load_async_to_lds_b128(
      (__attribute__((address_space(1))) i32x4*)gsrc,
      (__attribute__((address_space(3))) i32x4*)ldst, 0, 0);
#else
  *reinterpret_cast<uint4*>(ldst) = *reinterpret_cast<const uint4*>(gsrc);
#endif
}

__device__ __forceinline__ void wait_async_lds() {
#if USE_ASYNC_LDS
  __builtin_amdgcn_s_wait_asynccnt(0);
#endif
}

// ---------------------------------------------------------------------------
// Kernel 1: qkv = x @ Wqkv + bqkv ; write bf16 Q/K/V as [B][H][T][HD]
//   M = B*T = 8192, N = 3*D = 3072, K = D = 1024
//   tile 128x128x32, 256 threads = 8 waves (4 in M x 2 in N), 32x64 per wave
//   double-buffered LDS: stage k-step t+1 while WMMAs consume k-step t.
// ---------------------------------------------------------------------------
#define GM_BM 128
#define GM_BN 128
#define GM_BK 32
#define GM_LDA 40   // halfword stride (80B, 16B aligned, conflict-free)
#define GM_LDB 40

__global__ __launch_bounds__(256)
void qkv_gemm_kernel(const float* __restrict__ x,
                     const float* __restrict__ Wqkv,
                     const float* __restrict__ bqkv,
                     unsigned short* __restrict__ qws,
                     unsigned short* __restrict__ kws,
                     unsigned short* __restrict__ vws)
{
  __shared__ __align__(16) unsigned short sA[2][GM_BM * GM_LDA];   // [m][k]
  __shared__ __align__(16) unsigned short sB[2][GM_BN * GM_LDB];   // [n][k]

  const int tid  = threadIdx.x;
  const int wave = tid >> 5;
  const int lane = tid & 31;
  const int g    = lane >> 4;      // half-wave
  const int ln   = lane & 15;
  const int wm   = (wave & 3) * 32;    // wave M offset in tile
  const int wn   = (wave >> 2) * 64;   // wave N offset in tile
  const int m0   = blockIdx.y * GM_BM;
  const int n0   = blockIdx.x * GM_BN;

  auto stage = [&](int kt, int buf) {
    const int k0 = kt * GM_BK;
    // A: 128x32 fp32 -> bf16 (4 float4 per thread)
#pragma unroll
    for (int i = 0; i < 4; ++i) {
      const int idx = tid + 256 * i;
      const int row = idx >> 3;      // 8 float4 per row
      const int c4  = idx & 7;
      const float4 f = *reinterpret_cast<const float4*>(
          x + (size_t)(m0 + row) * D_ + k0 + c4 * 4);
      uint2 p;
      p.x = (unsigned int)f2bf(f.x) | ((unsigned int)f2bf(f.y) << 16);
      p.y = (unsigned int)f2bf(f.z) | ((unsigned int)f2bf(f.w) << 16);
      *reinterpret_cast<uint2*>(&sA[buf][row * GM_LDA + c4 * 4]) = p;
    }
    // B transposed: 32x128 fp32 -> bf16 [n][k]
#pragma unroll
    for (int i = 0; i < 4; ++i) {
      const int idx = tid + 256 * i;
      const int kr  = idx >> 5;      // 32 float4 per k-row
      const int c4  = idx & 31;
      const float4 f = *reinterpret_cast<const float4*>(
          Wqkv + (size_t)(k0 + kr) * (3 * D_) + n0 + c4 * 4);
      sB[buf][(c4 * 4 + 0) * GM_LDB + kr] = f2bf(f.x);
      sB[buf][(c4 * 4 + 1) * GM_LDB + kr] = f2bf(f.y);
      sB[buf][(c4 * 4 + 2) * GM_LDB + kr] = f2bf(f.z);
      sB[buf][(c4 * 4 + 3) * GM_LDB + kr] = f2bf(f.w);
    }
  };

  f32x8 acc[2][4];
#pragma unroll
  for (int mi = 0; mi < 2; ++mi)
#pragma unroll
    for (int ni = 0; ni < 4; ++ni) acc[mi][ni] = zero8();

  stage(0, 0);
  __syncthreads();

  constexpr int NK = D_ / GM_BK;
  for (int kt = 0; kt < NK; ++kt) {
    const int cur = kt & 1;

    Frag16 a[2], b[4];
#pragma unroll
    for (int mi = 0; mi < 2; ++mi) {
      const unsigned short* base = &sA[cur][(wm + mi * 16 + ln) * GM_LDA];
      a[mi].q[0] = *reinterpret_cast<const uint4*>(base + g * 16);
      a[mi].q[1] = *reinterpret_cast<const uint4*>(base + g * 16 + 8);
    }
#pragma unroll
    for (int ni = 0; ni < 4; ++ni) {
      const unsigned short* base = &sB[cur][(wn + ni * 16 + ln) * GM_LDB];
      b[ni].q[0] = *reinterpret_cast<const uint4*>(base + g * 16);
      b[ni].q[1] = *reinterpret_cast<const uint4*>(base + g * 16 + 8);
    }

    if (kt + 1 < NK) stage(kt + 1, cur ^ 1);   // overlap staging with WMMAs

#pragma unroll
    for (int mi = 0; mi < 2; ++mi)
#pragma unroll
      for (int ni = 0; ni < 4; ++ni)
        acc[mi][ni] = __builtin_amdgcn_wmma_f32_16x16x32_bf16(
            false, a[mi].v, false, b[ni].v, (short)0, acc[mi][ni], false, false);
    __syncthreads();
  }

  // ---- epilogue: split columns into per-head Q / K / V bf16 ----
#pragma unroll
  for (int mi = 0; mi < 2; ++mi)
#pragma unroll
    for (int ni = 0; ni < 4; ++ni)
#pragma unroll
      for (int r = 0; r < 8; ++r) {
        const int row = m0 + wm + mi * 16 + g * 8 + r;
        const int col = n0 + wn + ni * 16 + ln;
        const float v = acc[mi][ni][r] + bqkv[col];
        const int which = col >> 10;       // 0=Q 1=K 2=V
        const int rem   = col & 1023;
        const int h     = rem >> 6;
        const int hd    = rem & 63;
        const int bb    = row >> 11;       // / T_
        const int t     = row & 2047;
        unsigned short* dst = (which == 0) ? qws : (which == 1) ? kws : vws;
        dst[((((size_t)bb * H_ + h) * T_ + t) << 6) + hd] = f2bf(v);
      }
}

// ---------------------------------------------------------------------------
// Kernel 2: flash attention, one WG = (b, h, 128 query rows), 8 waves,
// each wave owns 16 query rows; key blocks of 64, causal bound 2*qt+2 blocks.
// K tiles staged with CDNA5 async global->LDS; V transposed through VGPRs.
// Double-buffered K/V tiles so block kb+1 streams in behind block kb's WMMAs.
// ---------------------------------------------------------------------------
#define AT_KLD 72   // padded LDS stride (144B rows, 16B aligned, bank-spread)
#define AT_PLD 72

__global__ __launch_bounds__(256)
void attn_kernel(const unsigned short* __restrict__ qws,
                 const unsigned short* __restrict__ kws,
                 const unsigned short* __restrict__ vws,
                 unsigned short* __restrict__ yatt)
{
  __shared__ __align__(16) unsigned short sK [2][64 * AT_KLD];     // [key][hd]
  __shared__ __align__(16) unsigned short sVt[2][64 * AT_KLD];     // [hd][key]
  __shared__ __align__(16) unsigned short sP [8 * 16 * AT_PLD];    // per-wave P

  const int tid  = threadIdx.x;
  const int wave = tid >> 5;
  const int lane = tid & 31;
  const int g    = lane >> 4;
  const int ln   = lane & 15;
  const int qt = blockIdx.x, h = blockIdx.y, b = blockIdx.z;
  const int qbase = qt * 128;

  const size_t headoff = (((size_t)b * H_ + h) * T_) << 6;   // * HD_
  const unsigned short* Q = qws + headoff;
  const unsigned short* K = kws + headoff;
  const unsigned short* V = vws + headoff;

  auto stageKV = [&](int kb, int buf) {
    const int kcol0 = kb * 64;
#pragma unroll
    for (int i = 0; i < 2; ++i) {
      const int idx = tid + 256 * i;
      const int row = idx >> 3;
      const int c8  = idx & 7;
      // K tile: straight bf16 copy -> async DMA into LDS (no VGPR round trip)
      copy_b128_to_lds(K + (size_t)(kcol0 + row) * HD_ + c8 * 8,
                       &sK[buf][row * AT_KLD + c8 * 8]);
      // V tile: needs transpose, goes through VGPRs
      U8x16 vv;
      vv.q = *reinterpret_cast<const uint4*>(
          V + (size_t)(kcol0 + row) * HD_ + c8 * 8);
#pragma unroll
      for (int j = 0; j < 8; ++j)
        sVt[buf][(c8 * 8 + j) * AT_KLD + row] = vv.u[j];
    }
  };

  // Q fragments for this wave's 16 rows (resident for the whole kernel)
  Frag16 aq[2];
  {
    const unsigned short* qrow = Q + (size_t)(qbase + wave * 16 + ln) * HD_;
#pragma unroll
    for (int ks = 0; ks < 2; ++ks) {
      aq[ks].q[0] = *reinterpret_cast<const uint4*>(qrow + ks * 32 + g * 16);
      aq[ks].q[1] = *reinterpret_cast<const uint4*>(qrow + ks * 32 + g * 16 + 8);
    }
  }

  f32x8 o[4];
#pragma unroll
  for (int c = 0; c < 4; ++c) o[c] = zero8();
  float mrow[8], lrow[8];
#pragma unroll
  for (int r = 0; r < 8; ++r) { mrow[r] = -3.0e38f; lrow[r] = 0.0f; }

  const int nkb = qt * 2 + 2;        // causal: keys up to qbase+127
  stageKV(0, 0);
  wait_async_lds();
  __syncthreads();

  for (int kb = 0; kb < nkb; ++kb) {
    const int cur   = kb & 1;
    const int kcol0 = kb * 64;

    // kick off next block's staging so the DMA overlaps this block's math
    if (kb + 1 < nkb) stageKV(kb + 1, cur ^ 1);

    // ---- S = Q K^T over 4 key sub-tiles ----
    f32x8 st[4];
#pragma unroll
    for (int c = 0; c < 4; ++c) {
      f32x8 s = zero8();
#pragma unroll
      for (int ks = 0; ks < 2; ++ks) {
        Frag16 bk;
        const unsigned short* base =
            &sK[cur][(c * 16 + ln) * AT_KLD + ks * 32 + g * 16];
        bk.q[0] = *reinterpret_cast<const uint4*>(base);
        bk.q[1] = *reinterpret_cast<const uint4*>(base + 8);
        s = __builtin_amdgcn_wmma_f32_16x16x32_bf16(
            false, aq[ks].v, false, bk.v, (short)0, s, false, false);
      }
      st[c] = s;
    }

    // ---- online softmax (rows live in this lane's half-wave) ----
    const int rowbase = qbase + wave * 16 + g * 8;
#pragma unroll
    for (int r = 0; r < 8; ++r) {
      const int row = rowbase + r;
      float xv[4];
      float mx = -3.0e38f;
#pragma unroll
      for (int c = 0; c < 4; ++c) {
        float xx = st[c][r] * 0.125f;                    // 1/sqrt(64)
        const int col = kcol0 + c * 16 + ln;
        if (col > row) xx += NEG_;                       // additive causal mask
        xv[c] = xx;
        mx = fmaxf(mx, xx);
      }
      mx = fmaxf(mx, __shfl_xor(mx, 1));
      mx = fmaxf(mx, __shfl_xor(mx, 2));
      mx = fmaxf(mx, __shfl_xor(mx, 4));
      mx = fmaxf(mx, __shfl_xor(mx, 8));
      const float mnew  = fmaxf(mrow[r], mx);
      const float alpha = __expf(mrow[r] - mnew);
      float sum = 0.0f;
#pragma unroll
      for (int c = 0; c < 4; ++c) {
        const float p = __expf(xv[c] - mnew);
        sum += p;
        sP[(wave * 16 + g * 8 + r) * AT_PLD + c * 16 + ln] = f2bf(p);
      }
      sum += __shfl_xor(sum, 1);
      sum += __shfl_xor(sum, 2);
      sum += __shfl_xor(sum, 4);
      sum += __shfl_xor(sum, 8);
      lrow[r] = lrow[r] * alpha + sum;
      mrow[r] = mnew;
#pragma unroll
      for (int c2 = 0; c2 < 4; ++c2) o[c2][r] *= alpha;
    }

    // ---- O += P V  (P re-read from own wave's LDS region; DS is in-order) ----
#pragma unroll
    for (int ks = 0; ks < 2; ++ks) {
      Frag16 ap;
      const unsigned short* pbase = &sP[(wave * 16 + ln) * AT_PLD + ks * 32];
      ap.q[0] = *reinterpret_cast<const uint4*>(pbase + g * 16);
      ap.q[1] = *reinterpret_cast<const uint4*>(pbase + g * 16 + 8);
#pragma unroll
      for (int c2 = 0; c2 < 4; ++c2) {
        Frag16 bv;
        const unsigned short* vb =
            &sVt[cur][(c2 * 16 + ln) * AT_KLD + ks * 32 + g * 16];
        bv.q[0] = *reinterpret_cast<const uint4*>(vb);
        bv.q[1] = *reinterpret_cast<const uint4*>(vb + 8);
        o[c2] = __builtin_amdgcn_wmma_f32_16x16x32_bf16(
            false, ap.v, false, bv.v, (short)0, o[c2], false, false);
      }
    }

    wait_async_lds();    // our async K loads into the other buffer are done
    __syncthreads();
  }

  // ---- normalize and emit yatt[b][t][h*64+hd] as bf16 ----
#pragma unroll
  for (int c2 = 0; c2 < 4; ++c2)
#pragma unroll
    for (int r = 0; r < 8; ++r) {
      const int row = qbase + wave * 16 + g * 8 + r;
      const float val = o[c2][r] / lrow[r];
      yatt[(((size_t)b * T_ + row) << 10) + (h << 6) + c2 * 16 + ln] = f2bf(val);
    }
}

// ---------------------------------------------------------------------------
// Kernel 3: out = yatt(bf16) @ Wout(f32->bf16) + bout, fp32 out.
//   M = 8192, N = 1024, K = 1024; same tiling as kernel 1, double-buffered,
//   A tile (already bf16) staged with async global->LDS DMA.
// ---------------------------------------------------------------------------
__global__ __launch_bounds__(256)
void out_gemm_kernel(const unsigned short* __restrict__ yatt,
                     const float* __restrict__ Wout,
                     const float* __restrict__ bout,
                     float* __restrict__ out)
{
  __shared__ __align__(16) unsigned short sA[2][GM_BM * GM_LDA];
  __shared__ __align__(16) unsigned short sB[2][GM_BN * GM_LDB];

  const int tid  = threadIdx.x;
  const int wave = tid >> 5;
  const int lane = tid & 31;
  const int g    = lane >> 4;
  const int ln   = lane & 15;
  const int wm   = (wave & 3) * 32;
  const int wn   = (wave >> 2) * 64;
  const int m0   = blockIdx.y * GM_BM;
  const int n0   = blockIdx.x * GM_BN;

  auto stage = [&](int kt, int buf) {
    const int k0 = kt * GM_BK;
    // A: already bf16 -> async DMA straight into LDS
#pragma unroll
    for (int i = 0; i < 2; ++i) {
      const int idx = tid + 256 * i;
      const int row = idx >> 2;        // 4 uint4 per row (32 halfs)
      const int c8  = idx & 3;
      copy_b128_to_lds(yatt + (size_t)(m0 + row) * D_ + k0 + c8 * 8,
                       &sA[buf][row * GM_LDA + c8 * 8]);
    }
    // B transposed with fp32->bf16 conversion
#pragma unroll
    for (int i = 0; i < 4; ++i) {
      const int idx = tid + 256 * i;
      const int kr  = idx >> 5;
      const int c4  = idx & 31;
      const float4 f = *reinterpret_cast<const float4*>(
          Wout + (size_t)(k0 + kr) * D_ + n0 + c4 * 4);
      sB[buf][(c4 * 4 + 0) * GM_LDB + kr] = f2bf(f.x);
      sB[buf][(c4 * 4 + 1) * GM_LDB + kr] = f2bf(f.y);
      sB[buf][(c4 * 4 + 2) * GM_LDB + kr] = f2bf(f.z);
      sB[buf][(c4 * 4 + 3) * GM_LDB + kr] = f2bf(f.w);
    }
  };

  f32x8 acc[2][4];
#pragma unroll
  for (int mi = 0; mi < 2; ++mi)
#pragma unroll
    for (int ni = 0; ni < 4; ++ni) acc[mi][ni] = zero8();

  stage(0, 0);
  wait_async_lds();
  __syncthreads();

  constexpr int NK = D_ / GM_BK;
  for (int kt = 0; kt < NK; ++kt) {
    const int cur = kt & 1;

    Frag16 a[2], bfr[4];
#pragma unroll
    for (int mi = 0; mi < 2; ++mi) {
      const unsigned short* base = &sA[cur][(wm + mi * 16 + ln) * GM_LDA];
      a[mi].q[0] = *reinterpret_cast<const uint4*>(base + g * 16);
      a[mi].q[1] = *reinterpret_cast<const uint4*>(base + g * 16 + 8);
    }
#pragma unroll
    for (int ni = 0; ni < 4; ++ni) {
      const unsigned short* base = &sB[cur][(wn + ni * 16 + ln) * GM_LDB];
      bfr[ni].q[0] = *reinterpret_cast<const uint4*>(base + g * 16);
      bfr[ni].q[1] = *reinterpret_cast<const uint4*>(base + g * 16 + 8);
    }

    if (kt + 1 < NK) stage(kt + 1, cur ^ 1);

#pragma unroll
    for (int mi = 0; mi < 2; ++mi)
#pragma unroll
      for (int ni = 0; ni < 4; ++ni)
        acc[mi][ni] = __builtin_amdgcn_wmma_f32_16x16x32_bf16(
            false, a[mi].v, false, bfr[ni].v, (short)0, acc[mi][ni], false, false);

    wait_async_lds();
    __syncthreads();
  }

#pragma unroll
  for (int mi = 0; mi < 2; ++mi)
#pragma unroll
    for (int ni = 0; ni < 4; ++ni)
#pragma unroll
      for (int r = 0; r < 8; ++r) {
        const int row = m0 + wm + mi * 16 + g * 8 + r;
        const int col = n0 + wn + ni * 16 + ln;
        out[(size_t)row * D_ + col] = acc[mi][ni][r] + bout[col];
      }
}

// ---------------------------------------------------------------------------
// Launcher. Inputs: x, attn_mask(unused: mask is exactly causal), Wqkv, bqkv,
// Wout, bout. Workspace: bf16 Q,K,V [B,H,T,64] + bf16 yatt [B,T,D] = 64 MB.
// ---------------------------------------------------------------------------
extern "C" void kernel_launch(void* const* d_in, const int* in_sizes, int n_in,
                              void* d_out, int out_size, void* d_ws, size_t ws_size,
                              hipStream_t stream) {
  const float* x    = (const float*)d_in[0];
  const float* Wqkv = (const float*)d_in[2];
  const float* bqkv = (const float*)d_in[3];
  const float* Wout = (const float*)d_in[4];
  const float* bout = (const float*)d_in[5];
  float* out = (float*)d_out;

  unsigned short* ws = (unsigned short*)d_ws;
  const size_t per = (size_t)B_ * H_ * T_ * HD_;   // 8M elements
  unsigned short* qws  = ws;
  unsigned short* kws  = ws + per;
  unsigned short* vws  = ws + 2 * per;
  unsigned short* yatt = ws + 3 * per;

  // 1) QKV projection: grid (N/128 = 24, M/128 = 64)
  qkv_gemm_kernel<<<dim3(24, 64), 256, 0, stream>>>(x, Wqkv, bqkv, qws, kws, vws);
  // 2) flash attention: grid (T/128 = 16, H = 16, B = 4)
  attn_kernel<<<dim3(16, 16, 4), 256, 0, stream>>>(qws, kws, vws, yatt);
  // 3) output projection: grid (N/128 = 8, M/128 = 64)
  out_gemm_kernel<<<dim3(8, 64), 256, 0, stream>>>(yatt, Wout, bout, out);
}
